// MultiHeadAttention_47820165874251
// MI455X (gfx1250) — matrix operand order repaired and checked by reading, not compiled, not run on hardware
//
#include <hip/hip_runtime.h>
#include <hip/hip_bf16.h>

// ---------------------------------------------------------------------------
// MI455X (gfx1250) fused MHA forward.  B=8,S=1024,D=1024,H=16,DH=64.
// Matrix math: v_wmma_f32_16x16x32_bf16 (wave32).  f32 -> bf16 at staging,
// f32 accumulate + f32 softmax.  k/v slabs staged LDS via
// global_load_async_to_lds_b128 (ASYNCcnt) when the builtin is available.
// ---------------------------------------------------------------------------

typedef __attribute__((ext_vector_type(16))) __bf16       v16bf;
typedef __attribute__((ext_vector_type(8)))  float        v8f;
typedef __attribute__((ext_vector_type(4)))  float        f32x4;
typedef __attribute__((ext_vector_type(4)))  unsigned int u32x4;

union BF16Frag {
    v16bf          v;
    u32x4          q[2];
    unsigned short u[16];
};

__device__ __forceinline__ unsigned short f2bf(float f) {
    unsigned int x = __float_as_uint(f);
    x += 0x7fffu + ((x >> 16) & 1u);      // round-to-nearest-even
    return (unsigned short)(x >> 16);
}

__device__ __forceinline__ v8f wmma_bf16(const v16bf& a, const v16bf& b, const v8f& c) {
    return __builtin_amdgcn_wmma_f32_16x16x32_bf16(false, a, false, b, (short)0, c, false, false);
}

// ---- CDNA5 async global->LDS copy (ASYNCcnt) -------------------------------
#if defined(__has_builtin)
#if __has_builtin(__builtin_amdgcn_global_load_async_to_lds_b128) && \
    __has_builtin(__builtin_amdgcn_s_wait_asynccnt)
#define HAVE_ASYNC_LDS 1
#endif
#endif

#if HAVE_ASYNC_LDS
// prototype (from hipcc diagnostic): (V4i AS1*, V4i AS3*, imm int, imm int)
typedef int v4i __attribute__((vector_size(4 * sizeof(int))));
__device__ __forceinline__ void async_cp16(const void* g, void* l) {
    __builtin_amdgcn_global_load_async_to_lds_b128(
        (__attribute__((address_space(1))) v4i*)(g),
        (__attribute__((address_space(3))) v4i*)(l), 0, 0);
}
#endif

// ===========================================================================
// proj_gemm<BF_OUT>: C[8192,1024] = A[8192,1024](f32) @ B[1024,1024](f32)
// 128x128 block tile, 8 waves x (2 row-tiles x 4 col-tiles), K chunk 32.
// ===========================================================================
#define GK      1024
#define MTILE   128
#define NTILE   128
#define KC      32
#define ASTR    40   // u16 stride (80B rows -> 16B-aligned 8-elem runs)
#define BSTR    40

template <bool BF_OUT>
__global__ __launch_bounds__(256)
void proj_gemm(const float* __restrict__ A, const float* __restrict__ B,
               void* __restrict__ Cout)
{
    __shared__ unsigned short As[MTILE * ASTR];   // [m][k]
    __shared__ unsigned short Bs[NTILE * BSTR];   // transposed: [n][k]

    const int n0   = blockIdx.x * NTILE;
    const int m0   = blockIdx.y * MTILE;
    const int tid  = threadIdx.x;
    const int lane = tid & 31;
    const int wv   = tid >> 5;
    const int lm   = lane & 15;
    const int hl   = lane >> 4;

    const int rtb = (wv & 3) * 2;   // row-tile base: 0,2,4,6
    const int ctb = (wv >> 2) * 4;  // col-tile base: 0,4

    // per-thread staging coordinates
    const int ar = tid >> 1, ac = (tid & 1) * 16;     // A: row, 16-col chunk
    const int bk = tid >> 3, bn = (tid & 7) * 16;     // B: k row, 16-n chunk
    const float* asrc = A + (size_t)(m0 + ar) * GK + ac;
    const float* bsrc = B + (size_t)bk * GK + n0 + bn;

    v8f acc[2][4];
    for (int i = 0; i < 2; ++i)
        for (int j = 0; j < 4; ++j)
            for (int e = 0; e < 8; ++e) acc[i][j][e] = 0.f;

    for (int kb = 0; kb < GK; kb += KC) {
        __syncthreads();
        if (kb + KC < GK) {                    // prefetch next chunk
            __builtin_prefetch(asrc + kb + KC, 0, 0);
            __builtin_prefetch(bsrc + (size_t)(kb + KC) * GK, 0, 0);
        }
        {   // stage A tile (f32 -> bf16), two 16B LDS stores per thread
            const float* s = asrc + kb;
            f32x4 f0 = *(const f32x4*)(s);
            f32x4 f1 = *(const f32x4*)(s + 4);
            f32x4 f2 = *(const f32x4*)(s + 8);
            f32x4 f3 = *(const f32x4*)(s + 12);
            union { u32x4 q[2]; unsigned short u[16]; } pk;
            #pragma unroll
            for (int j = 0; j < 4; ++j) {
                pk.u[j]      = f2bf(f0[j]);
                pk.u[j + 4]  = f2bf(f1[j]);
                pk.u[j + 8]  = f2bf(f2[j]);
                pk.u[j + 12] = f2bf(f3[j]);
            }
            *(u32x4*)(As + ar * ASTR + ac)     = pk.q[0];
            *(u32x4*)(As + ar * ASTR + ac + 8) = pk.q[1];
        }
        {   // stage B tile transposed: vector read, scatter u16 to [n][k]
            const float* s = bsrc + (size_t)kb * GK;
            f32x4 f0 = *(const f32x4*)(s);
            f32x4 f1 = *(const f32x4*)(s + 4);
            f32x4 f2 = *(const f32x4*)(s + 8);
            f32x4 f3 = *(const f32x4*)(s + 12);
            #pragma unroll
            for (int j = 0; j < 4; ++j) {
                Bs[(bn + j)      * BSTR + bk] = f2bf(f0[j]);
                Bs[(bn + j + 4)  * BSTR + bk] = f2bf(f1[j]);
                Bs[(bn + j + 8)  * BSTR + bk] = f2bf(f2[j]);
                Bs[(bn + j + 12) * BSTR + bk] = f2bf(f3[j]);
            }
        }
        __syncthreads();

        BF16Frag a[2], b[4];
        #pragma unroll
        for (int i = 0; i < 2; ++i) {   // A layout: 8-elem runs @8*hl, 16+8*hl
            const unsigned short* p = As + ((rtb + i) * 16 + lm) * ASTR;
            a[i].q[0] = *(const u32x4*)(p + 8 * hl);
            a[i].q[1] = *(const u32x4*)(p + 16 + 8 * hl);
        }
        #pragma unroll
        for (int j = 0; j < 4; ++j) {   // B layout: 8-elem runs @8*hl, 16+8*hl
            const unsigned short* p = Bs + ((ctb + j) * 16 + lm) * BSTR + 8 * hl;
            b[j].q[0] = *(const u32x4*)(p);
            b[j].q[1] = *(const u32x4*)(p + 16);
        }
        #pragma unroll
        for (int i = 0; i < 2; ++i)
            #pragma unroll
            for (int j = 0; j < 4; ++j)
                acc[i][j] = wmma_bf16(a[i].v, b[j].v, acc[i][j]);
    }

    // epilogue (branch-free, type chosen at compile time)
    #pragma unroll
    for (int i = 0; i < 2; ++i)
        #pragma unroll
        for (int j = 0; j < 4; ++j)
            #pragma unroll
            for (int e = 0; e < 8; ++e) {
                const int m = m0 + (rtb + i) * 16 + e + 8 * hl;
                const int n = n0 + (ctb + j) * 16 + lm;
                const size_t idx = (size_t)m * GK + n;
                const float val = acc[i][j][e];
                if (BF_OUT) ((unsigned short*)Cout)[idx] = f2bf(val);
                else        ((float*)Cout)[idx]          = val;
            }
}

// ===========================================================================
// attn_kernel: per (g in [0,128), 16-row block) — S = q@k^T *0.125 masked,
// row softmax, att -> global f32, P(bf16) @ v + residual, relu -> out.
// q/k/v per g are contiguous [1024,64] bf16 slabs at offset g*65536.
// k_g is async-staged to LDS; the same LDS buffer is then async-refilled
// with v_g while softmax runs (copy/compute overlap on ASYNCcnt).
// ===========================================================================
#define SSTR 1028   // f32 stride (bank-skewed)
#define PSTR 1040   // u16 stride (2080B rows -> 16B-aligned runs)
#define KVSTR 72    // u16 stride for staged [1024,64] slab (144B rows)

__global__ __launch_bounds__(256)
void attn_kernel(const unsigned short* __restrict__ Qb,
                 const unsigned short* __restrict__ Kb,
                 const unsigned short* __restrict__ Vb,
                 const float* __restrict__ RW,
                 float* __restrict__ att, float* __restrict__ out)
{
    __shared__ float          S[16 * SSTR];
    __shared__ unsigned short P[16 * PSTR];
    __shared__ float          rowM[16], rowSum[16];
#if HAVE_ASYNC_LDS
    __shared__ unsigned short KV[1024 * KVSTR];   // k_g, later reused for v_g
#endif

    const int g    = blockIdx.y;
    const int r0   = blockIdx.x * 16;
    const int tid  = threadIdx.x;
    const int lane = tid & 31;
    const int wv   = tid >> 5;
    const int lm   = lane & 15;
    const int hl   = lane >> 4;
    const size_t gbase = (size_t)g * 65536;   // g * 1024 * 64

#if HAVE_ASYNC_LDS
    {   // async stage k_g [1024,64] -> LDS (pad to KVSTR); 32 x b128 / thread
        const unsigned short* gsrc = Kb + gbase;
        #pragma unroll 4
        for (int i = 0; i < 32; ++i) {
            const int idx = tid + 256 * i;          // 16B-chunk id
            const int n = idx >> 3, c = idx & 7;
            async_cp16(gsrc + n * 64 + c * 8, KV + n * KVSTR + c * 8);
        }
    }
#endif

    // ---- q block fragments (overlap with async K copy) ----
    BF16Frag aq[2];
    {
        const unsigned short* qrow = Qb + gbase + (size_t)(r0 + lm) * 64;
        #pragma unroll
        for (int s = 0; s < 2; ++s) {
            aq[s].q[0] = *(const u32x4*)(qrow + s * 32 + 8 * hl);
            aq[s].q[1] = *(const u32x4*)(qrow + s * 32 + 16 + 8 * hl);
        }
    }

#if HAVE_ASYNC_LDS
    __builtin_amdgcn_s_wait_asynccnt(0);
    __syncthreads();
#endif

    // ---- S = q @ k^T (M=16, N=1024, K=64): 8 waves x 8 col-tiles x 2 ksteps
    #pragma unroll
    for (int t = 0; t < 8; ++t) {
        const int ct = wv * 8 + t;            // col tile 0..63
        v8f acc;
        for (int e = 0; e < 8; ++e) acc[e] = 0.f;
        #pragma unroll
        for (int s = 0; s < 2; ++s) {
            BF16Frag bk;                       // B = k^T: lane n = ct*16+lm
#if HAVE_ASYNC_LDS
            const unsigned short* kr = KV + (ct * 16 + lm) * KVSTR + s * 32 + 8 * hl;
#else
            const unsigned short* kr = Kb + gbase + (size_t)(ct * 16 + lm) * 64
                                       + s * 32 + 8 * hl;
#endif
            bk.q[0] = *(const u32x4*)(kr);
            bk.q[1] = *(const u32x4*)(kr + 16);
            acc = wmma_bf16(aq[s].v, bk.v, acc);
        }
        #pragma unroll
        for (int e = 0; e < 8; ++e) {          // scale + strict-upper mask
            const int r = e + 8 * hl;
            const int c = ct * 16 + lm;
            S[r * SSTR + c] = (c > r0 + r) ? acc[e] * 0.125f : 0.0f;
        }
    }
    __syncthreads();                            // K slab no longer needed

#if HAVE_ASYNC_LDS
    {   // async refill same LDS with v_g; overlaps softmax below
        const unsigned short* gsrc = Vb + gbase;
        #pragma unroll 4
        for (int i = 0; i < 32; ++i) {
            const int idx = tid + 256 * i;
            const int n = idx >> 3, c = idx & 7;
            async_cp16(gsrc + n * 64 + c * 8, KV + n * KVSTR + c * 8);
        }
    }
#endif

    // ---- softmax stats: wave wv owns rows 2wv, 2wv+1
    #pragma unroll
    for (int rr = 0; rr < 2; ++rr) {
        const int r = wv * 2 + rr;
        float m = -1e30f;
        for (int c = lane; c < 1024; c += 32) m = fmaxf(m, S[r * SSTR + c]);
        for (int off = 16; off > 0; off >>= 1) m = fmaxf(m, __shfl_xor(m, off, 32));
        float sum = 0.f;
        for (int c = lane; c < 1024; c += 32) sum += __expf(S[r * SSTR + c] - m);
        for (int off = 16; off > 0; off >>= 1) sum += __shfl_xor(sum, off, 32);
        if (lane == 0) { rowM[r] = m; rowSum[r] = sum; }
    }
    __syncthreads();

    // ---- normalize: att (f32x4, coalesced) to global + P (bf16) to LDS
    {
        float* attg = att + (size_t)g * 1048576 + (size_t)r0 * 1024;
        #pragma unroll 2
        for (int i = 0; i < 16; ++i) {
            const int e = (tid + 256 * i) * 4;
            const int r = e >> 10, c = e & 1023;
            const float inv = 1.0f / rowSum[r];
            f32x4 pv;
            #pragma unroll
            for (int j = 0; j < 4; ++j)
                pv[j] = __expf(S[r * SSTR + c + j] - rowM[r]) * inv;
            *(f32x4*)(attg + (size_t)r * 1024 + c) = pv;
            union { unsigned short u[4]; unsigned long long q; } pb;
            #pragma unroll
            for (int j = 0; j < 4; ++j) pb.u[j] = f2bf(pv[j]);
            *(unsigned long long*)(P + r * PSTR + c) = pb.q;
        }
    }
#if HAVE_ASYNC_LDS
    __builtin_amdgcn_s_wait_asynccnt(0);        // v_g landed (own wave's cnt)
#endif
    __syncthreads();                            // + everyone else's

    // ---- out = P(16x1024) @ v(1024x64): 4 waves, 1 tile each, 32 ksteps
    if (wv < 4) {
        const int ct = wv;                     // dh tile 0..3
        v8f acc;
        for (int e = 0; e < 8; ++e) acc[e] = 0.f;
        for (int kb = 0; kb < 1024; kb += 32) {
            BF16Frag ap, bv;
            const unsigned short* pr = P + lm * PSTR + kb;
            ap.q[0] = *(const u32x4*)(pr + 8 * hl);
            ap.q[1] = *(const u32x4*)(pr + 16 + 8 * hl);
            const int n = ct * 16 + lm;
#if HAVE_ASYNC_LDS
            const unsigned short* vs = KV + (kb + 8 * hl) * KVSTR + n;
            #pragma unroll
            for (int j = 0; j < 16; ++j) bv.u[j] = vs[(j + (j & 8)) * KVSTR];
#else
            const unsigned short* vs = Vb + gbase + (size_t)(kb + 8 * hl) * 64 + n;
            #pragma unroll
            for (int j = 0; j < 16; ++j) bv.u[j] = vs[(size_t)(j + (j & 8)) * 64];
#endif
            acc = wmma_bf16(ap.v, bv.v, acc);
        }
        #pragma unroll
        for (int e = 0; e < 8; ++e) {
            const int r = e + 8 * hl;
            const int n = ct * 16 + lm;
            const size_t idx = gbase + (size_t)(r0 + r) * 64 + n;
            out[idx] = fmaxf(acc[e] + RW[idx], 0.f);
        }
    }
}

// ===========================================================================
extern "C" void kernel_launch(void* const* d_in, const int* in_sizes, int n_in,
                              void* d_out, int out_size, void* d_ws, size_t ws_size,
                              hipStream_t stream) {
    (void)in_sizes; (void)n_in; (void)out_size; (void)ws_size;

    const float* q  = (const float*)d_in[0];
    const float* k  = (const float*)d_in[1];
    const float* v  = (const float*)d_in[2];
    const float* Wq = (const float*)d_in[3];
    const float* Wk = (const float*)d_in[4];
    const float* Wv = (const float*)d_in[5];
    const float* Wr = (const float*)d_in[6];
    // d_in[7] = num_heads (compile-time 16 here)

    // workspace: Qb/Kb/Vb bf16 [8192x1024] + residual f32 [8192x1024] = 84MB
    unsigned short* Qb = (unsigned short*)d_ws;
    unsigned short* Kb = Qb + (size_t)8192 * 1024;
    unsigned short* Vb = Kb + (size_t)8192 * 1024;
    float*          RW = (float*)(Vb + (size_t)8192 * 1024);

    float* outp = (float*)d_out;                    // [8,1024,1024]
    float* attp = outp + (size_t)8 * 1024 * 1024;   // [128,1024,1024]

    dim3 ggrid(8, 64);                              // (N/128, M/128)
    proj_gemm<true ><<<ggrid, 256, 0, stream>>>(q, Wq, Qb);
    proj_gemm<true ><<<ggrid, 256, 0, stream>>>(k, Wk, Kb);
    proj_gemm<true ><<<ggrid, 256, 0, stream>>>(v, Wv, Vb);
    proj_gemm<false><<<ggrid, 256, 0, stream>>>(q, Wr, RW);

    dim3 agrid(64, 128);                            // (row-blocks, g)
    attn_kernel<<<agrid, 256, 0, stream>>>(Qb, Kb, Vb, RW, attp, outp);
}
